// MoELayer_44968307589634
// MI455X (gfx1250) — matrix operand order repaired and checked by reading, compiled, not run on hardware
//
#include <hip/hip_runtime.h>
#include <hip/hip_bf16.h>

// MoE (LN + top-2 router + GLU-FFN experts) for MI455X / gfx1250.
// Strategy: token gather per expert (4x less FLOP than reference's dense form),
// bf16 WMMA (v_wmma_f32_16x16x32_bf16) with fp32 accumulation; weights repacked
// per call into a B-operand lane layout (16 contiguous K per lane half) so GEMM inner loops are pure
// contiguous b128 loads (bf16 weights ~100MB -> resident in 192MB L2).
// GEMM waves use 2Mx4N register blocking: 8 WMMAs per 12 b128 loads to keep
// the matrix pipe fed.  Workspace requirement: ~139 MB.

namespace {

constexpr int kD    = 1024;   // model dim
constexpr int kE    = 8;      // experts
constexpr int kH    = 2048;   // hidden
constexpr int kTopK = 2;
constexpr int kN    = 2 * 1024;        // B*S tokens
constexpr int kRows = kN * kTopK;      // 4096 compacted (token,expert) rows
constexpr float kLnEps = 1e-5f;

typedef __attribute__((ext_vector_type(16))) __bf16 bf16x16;
typedef __attribute__((ext_vector_type(8)))  float  f32x8;

__device__ __forceinline__ unsigned short f2bf(float f) {
  unsigned int u = __float_as_uint(f);
  u += 0x7FFFu + ((u >> 16) & 1u);    // round-to-nearest-even
  return (unsigned short)(u >> 16);
}

union FragU { uint4 q[2]; bf16x16 v; };

// A operand, 16-bit 16x32: lane-half 0 holds K[0..15],
// half 1 holds K[16..31] (matches packed B order) -> two 16B segments per lane.
__device__ __forceinline__ bf16x16 load_a_frag(const unsigned short* rowk, int half) {
  FragU u;
  u.q[0] = *reinterpret_cast<const uint4*>(rowk + half * 16);
  u.q[1] = *reinterpret_cast<const uint4*>(rowk + half * 16 + 8);
  return u.v;
}

// B operand pre-packed: lane holds one column, 16 contiguous K values (32B).
__device__ __forceinline__ bf16x16 load_b_frag(const unsigned short* p) {
  FragU u;
  u.q[0] = reinterpret_cast<const uint4*>(p)[0];
  u.q[1] = reinterpret_cast<const uint4*>(p)[1];
  return u.v;
}

__device__ __forceinline__ float gelu_exact(float g) {
  return 0.5f * g * (1.0f + erff(g * 0.70710678118654752f));
}

#define WMMA_BF16(A, B, C) \
  __builtin_amdgcn_wmma_f32_16x16x32_bf16(false, (A), false, (B), (short)0, (C), false, false)

__global__ void k_init(int* counts) {
  if (threadIdx.x < kE) counts[threadIdx.x] = 0;
}

// One block per token: LayerNorm, xn->bf16, router logits/softmax/top2, mask.
__global__ __launch_bounds__(256)
void k_ln_router(const float* __restrict__ x, const float* __restrict__ ln_w,
                 const float* __restrict__ ln_b, const float* __restrict__ gate_w,
                 unsigned short* __restrict__ xn_bf, float* __restrict__ mask_out,
                 int* __restrict__ counts, int* __restrict__ expert_sel,
                 float* __restrict__ expert_wt) {
  __shared__ float red[256];
  __shared__ float logits[kE];
  const int t = blockIdx.x, tid = threadIdx.x;
  const float4 v = reinterpret_cast<const float4*>(x + (size_t)t * kD)[tid];

  red[tid] = v.x + v.y + v.z + v.w;
  __syncthreads();
  for (int o = 128; o > 0; o >>= 1) { if (tid < o) red[tid] += red[tid + o]; __syncthreads(); }
  const float mu = red[0] * (1.0f / kD);
  __syncthreads();

  const float d0 = v.x - mu, d1 = v.y - mu, d2 = v.z - mu, d3 = v.w - mu;
  red[tid] = d0 * d0 + d1 * d1 + d2 * d2 + d3 * d3;
  __syncthreads();
  for (int o = 128; o > 0; o >>= 1) { if (tid < o) red[tid] += red[tid + o]; __syncthreads(); }
  const float rstd = rsqrtf(red[0] * (1.0f / kD) + kLnEps);
  __syncthreads();

  const float4 w4 = reinterpret_cast<const float4*>(ln_w)[tid];
  const float4 b4 = reinterpret_cast<const float4*>(ln_b)[tid];
  const float n0 = d0 * rstd * w4.x + b4.x;
  const float n1 = d1 * rstd * w4.y + b4.y;
  const float n2 = d2 * rstd * w4.z + b4.z;
  const float n3 = d3 * rstd * w4.w + b4.w;

  uint2 packed;
  packed.x = (unsigned)f2bf(n0) | ((unsigned)f2bf(n1) << 16);
  packed.y = (unsigned)f2bf(n2) | ((unsigned)f2bf(n3) << 16);
  reinterpret_cast<uint2*>(xn_bf + (size_t)t * kD)[tid] = packed;

  for (int e = 0; e < kE; ++e) {
    const float4 g = reinterpret_cast<const float4*>(gate_w + (size_t)e * kD)[tid];
    red[tid] = n0 * g.x + n1 * g.y + n2 * g.z + n3 * g.w;
    __syncthreads();
    for (int o = 128; o > 0; o >>= 1) { if (tid < o) red[tid] += red[tid + o]; __syncthreads(); }
    if (tid == 0) logits[e] = red[0];
    __syncthreads();
  }

  if (tid == 0) {
    float w[kE], mx = logits[0];
    for (int e = 1; e < kE; ++e) mx = fmaxf(mx, logits[e]);
    float sum = 0.f;
    for (int e = 0; e < kE; ++e) { w[e] = expf(logits[e] - mx); sum += w[e]; }
    const float inv = 1.0f / sum;
    for (int e = 0; e < kE; ++e) w[e] *= inv;
    int i0 = 0;
    for (int e = 1; e < kE; ++e) if (w[e] > w[i0]) i0 = e;
    int i1 = (i0 == 0) ? 1 : 0;
    for (int e = 0; e < kE; ++e) if (e != i0 && w[e] > w[i1]) i1 = e;
    const float rn = 1.0f / (w[i0] + w[i1]);
    const float g0 = w[i0] * rn, g1 = w[i1] * rn;
    float* m = mask_out + (size_t)t * kE;
    for (int e = 0; e < kE; ++e) m[e] = 0.f;
    m[i0] = g0; m[i1] = g1;
    expert_sel[2 * t] = i0;  expert_sel[2 * t + 1] = i1;
    expert_wt[2 * t] = g0;   expert_wt[2 * t + 1] = g1;
    atomicAdd(&counts[i0], 1);
    atomicAdd(&counts[i1], 1);
  }
}

__global__ void k_offsets(const int* __restrict__ counts, int* __restrict__ offsets,
                          int* __restrict__ cursors) {
  if (threadIdx.x == 0) {
    int acc = 0;
    for (int e = 0; e < kE; ++e) { offsets[e] = acc; cursors[e] = acc; acc += counts[e]; }
    offsets[kE] = acc;   // == kRows
  }
}

__global__ __launch_bounds__(256)
void k_scatter(const int* __restrict__ expert_sel, const float* __restrict__ expert_wt,
               int* __restrict__ cursors, int* __restrict__ row_token,
               float* __restrict__ row_gate, int* __restrict__ token_slots) {
  const int t = blockIdx.x * blockDim.x + threadIdx.x;
  if (t >= kN) return;
  for (int j = 0; j < kTopK; ++j) {
    const int e = expert_sel[2 * t + j];
    const int slot = atomicAdd(&cursors[e], 1);
    row_token[slot] = t;
    row_gate[slot] = expert_wt[2 * t + j];
    token_slots[2 * t + j] = slot;
  }
}

// Repack fp32 [E][Kdim][Ncols] row-major -> bf16 B-operand tiles:
// dst[e][ntile][kc][lane][16]; lane&15 = column, lane>>4 picks K half.
__global__ __launch_bounds__(256)
void k_pack(const float* __restrict__ src, unsigned short* __restrict__ dst,
            int Kdim, int Ncols, int ntiles, int kcs, long total) {
  const long gid = (long)blockIdx.x * blockDim.x + threadIdx.x;
  if (gid >= total) return;
  const int lane = (int)(gid & 31);
  long q = gid >> 5;
  const int kc = (int)(q % kcs); q /= kcs;
  const int nt = (int)(q % ntiles);
  const int e  = (int)(q / ntiles);
  const int n  = nt * 16 + (lane & 15);
  const int h  = lane >> 4;
  const float* s = src + ((size_t)e * Kdim + (size_t)kc * 32 + h * 16) * Ncols + n;
  unsigned int u[8];
#pragma unroll
  for (int tp = 0; tp < 8; ++tp) {
    const unsigned lo = f2bf(s[(size_t)(2 * tp) * Ncols]);
    const unsigned hi = f2bf(s[(size_t)(2 * tp + 1) * Ncols]);
    u[tp] = lo | (hi << 16);
  }
  unsigned short* d = dst + ((size_t)(e * ntiles + nt) * kcs + kc) * 512 + lane * 16;
  uint4* d4 = reinterpret_cast<uint4*>(d);
  d4[0] = make_uint4(u[0], u[1], u[2], u[3]);
  d4[1] = make_uint4(u[4], u[5], u[6], u[7]);
}

// Stage 1: act[rows, H] = gelu(xn @ w1_gate) * (xn @ w1_x), gathered per expert.
// 2Mx4N register blocking: 2 m-frags x 2 (x,gate) col pairs = 8 WMMAs / K-chunk.
__global__ __launch_bounds__(256)
void k_ffn1(const unsigned short* __restrict__ xn_bf, const unsigned short* __restrict__ pw1,
            const int* __restrict__ counts, const int* __restrict__ offsets,
            const int* __restrict__ row_token, unsigned short* __restrict__ act) {
  const int e = blockIdx.z;
  const int cnt = counts[e];
  const int mstart = blockIdx.y * 32;
  if (mstart >= cnt) return;                 // block-uniform: EXEC stays all-1s
  const int offs = offsets[e];
  const int lane = threadIdx.x & 31;
  const int wave = threadIdx.x >> 5;
  const int half = lane >> 4;
  const int nlo  = lane & 15;

  int mi0 = mstart + nlo;
  int mi1 = mstart + 16 + nlo;
  if (mi0 >= cnt) mi0 = cnt - 1;             // clamp padding rows to a valid row
  if (mi1 >= cnt) mi1 = cnt - 1;
  const unsigned short* arow0 = xn_bf + (size_t)row_token[offs + mi0] * kD;
  const unsigned short* arow1 = xn_bf + (size_t)row_token[offs + mi1] * kD;

  const int ntx0 = blockIdx.x * 16 + wave * 2;   // x-part col tiles [0, H/16)
  const int ntx1 = ntx0 + 1;
  const size_t tstep = (size_t)(kD / 32) * 512;  // packed elements per n-tile
  const unsigned short* bx0 = pw1 + (size_t)(e * (2 * kH / 16) + ntx0) * tstep + lane * 16;
  const unsigned short* bx1 = pw1 + (size_t)(e * (2 * kH / 16) + ntx1) * tstep + lane * 16;
  const unsigned short* bg0 = bx0 + (size_t)(kH / 16) * tstep;   // gate-part tiles
  const unsigned short* bg1 = bx1 + (size_t)(kH / 16) * tstep;

  f32x8 ax00 = {}, ag00 = {}, ax01 = {}, ag01 = {};
  f32x8 ax10 = {}, ag10 = {}, ax11 = {}, ag11 = {};
#pragma unroll 2
  for (int kc = 0; kc < kD / 32; ++kc) {
    const bf16x16 a0 = load_a_frag(arow0 + kc * 32, half);
    const bf16x16 a1 = load_a_frag(arow1 + kc * 32, half);
    const bf16x16 b0 = load_b_frag(bx0 + (size_t)kc * 512);
    const bf16x16 b1 = load_b_frag(bx1 + (size_t)kc * 512);
    const bf16x16 b2 = load_b_frag(bg0 + (size_t)kc * 512);
    const bf16x16 b3 = load_b_frag(bg1 + (size_t)kc * 512);
    ax00 = WMMA_BF16(a0, b0, ax00);
    ax01 = WMMA_BF16(a0, b1, ax01);
    ag00 = WMMA_BF16(a0, b2, ag00);
    ag01 = WMMA_BF16(a0, b3, ag01);
    ax10 = WMMA_BF16(a1, b0, ax10);
    ax11 = WMMA_BF16(a1, b1, ax11);
    ag10 = WMMA_BF16(a1, b2, ag10);
    ag11 = WMMA_BF16(a1, b3, ag11);
  }

  const int col0 = ntx0 * 16;
  const int col1 = ntx1 * 16;
#pragma unroll
  for (int r = 0; r < 8; ++r) {
    const int m = r + half * 8;              // C/D layout: M = r + 8*(lane>>4)
    const int gm0 = mstart + m;
    if (gm0 < cnt) {
      unsigned short* arow = act + (size_t)(offs + gm0) * kH;
      arow[col0 + nlo] = f2bf(gelu_exact(ag00[r]) * ax00[r]);
      arow[col1 + nlo] = f2bf(gelu_exact(ag01[r]) * ax01[r]);
    }
    const int gm1 = mstart + 16 + m;
    if (gm1 < cnt) {
      unsigned short* arow = act + (size_t)(offs + gm1) * kH;
      arow[col0 + nlo] = f2bf(gelu_exact(ag10[r]) * ax10[r]);
      arow[col1 + nlo] = f2bf(gelu_exact(ag11[r]) * ax11[r]);
    }
  }
}

// Stage 2: eo[slot, D] = gate[slot] * (act[slot] @ w2).
// 2Mx4N register blocking: 2 m-frags x 4 adjacent n-tiles = 8 WMMAs / K-chunk.
__global__ __launch_bounds__(256)
void k_ffn2(const unsigned short* __restrict__ act, const unsigned short* __restrict__ pw2,
            const int* __restrict__ counts, const int* __restrict__ offsets,
            const float* __restrict__ row_gate, float* __restrict__ eo) {
  const int e = blockIdx.z;
  const int cnt = counts[e];
  const int mstart = blockIdx.y * 32;
  if (mstart >= cnt) return;
  const int offs = offsets[e];
  const int lane = threadIdx.x & 31;
  const int wave = threadIdx.x >> 5;
  const int half = lane >> 4;
  const int nlo  = lane & 15;

  int mi0 = mstart + nlo;
  int mi1 = mstart + 16 + nlo;
  if (mi0 >= cnt) mi0 = cnt - 1;
  if (mi1 >= cnt) mi1 = cnt - 1;
  const unsigned short* arow0 = act + (size_t)(offs + mi0) * kH;
  const unsigned short* arow1 = act + (size_t)(offs + mi1) * kH;

  const int nt0 = blockIdx.x * 32 + wave * 4;   // over D/16, 4 tiles per wave
  const size_t tstep = (size_t)(kH / 32) * 512;
  const unsigned short* bw0 = pw2 + (size_t)(e * (kD / 16) + nt0) * tstep + lane * 16;
  const unsigned short* bw1 = bw0 + tstep;
  const unsigned short* bw2 = bw1 + tstep;
  const unsigned short* bw3 = bw2 + tstep;

  f32x8 a00 = {}, a01 = {}, a02 = {}, a03 = {};
  f32x8 a10 = {}, a11 = {}, a12 = {}, a13 = {};
#pragma unroll 2
  for (int kc = 0; kc < kH / 32; ++kc) {
    const bf16x16 a0 = load_a_frag(arow0 + kc * 32, half);
    const bf16x16 a1 = load_a_frag(arow1 + kc * 32, half);
    const bf16x16 b0 = load_b_frag(bw0 + (size_t)kc * 512);
    const bf16x16 b1 = load_b_frag(bw1 + (size_t)kc * 512);
    const bf16x16 b2 = load_b_frag(bw2 + (size_t)kc * 512);
    const bf16x16 b3 = load_b_frag(bw3 + (size_t)kc * 512);
    a00 = WMMA_BF16(a0, b0, a00);
    a01 = WMMA_BF16(a0, b1, a01);
    a02 = WMMA_BF16(a0, b2, a02);
    a03 = WMMA_BF16(a0, b3, a03);
    a10 = WMMA_BF16(a1, b0, a10);
    a11 = WMMA_BF16(a1, b1, a11);
    a12 = WMMA_BF16(a1, b2, a12);
    a13 = WMMA_BF16(a1, b3, a13);
  }

  const int col = nt0 * 16 + nlo;
#pragma unroll
  for (int r = 0; r < 8; ++r) {
    const int m = r + half * 8;
    const int gm0 = mstart + m;
    if (gm0 < cnt) {
      const int slot = offs + gm0;
      const float g = row_gate[slot];
      float* orow = eo + (size_t)slot * kD + col;
      orow[0]  = a00[r] * g;
      orow[16] = a01[r] * g;
      orow[32] = a02[r] * g;
      orow[48] = a03[r] * g;
    }
    const int gm1 = mstart + 16 + m;
    if (gm1 < cnt) {
      const int slot = offs + gm1;
      const float g = row_gate[slot];
      float* orow = eo + (size_t)slot * kD + col;
      orow[0]  = a10[r] * g;
      orow[16] = a11[r] * g;
      orow[32] = a12[r] * g;
      orow[48] = a13[r] * g;
    }
  }
}

// Residual + top-2 combine (float4 per thread).
__global__ __launch_bounds__(256)
void k_final(const float* __restrict__ x, const float* __restrict__ eo,
             const int* __restrict__ token_slots, float* __restrict__ out) {
  const int idx = blockIdx.x * 256 + threadIdx.x;   // < kN*kD/4
  const int t  = idx >> 8;                          // kD/4 == 256 float4 per row
  const int d4 = idx & 255;
  const int s0 = token_slots[2 * t], s1 = token_slots[2 * t + 1];
  const float4 xv = reinterpret_cast<const float4*>(x)[idx];
  const float4 e0 = reinterpret_cast<const float4*>(eo + (size_t)s0 * kD)[d4];
  const float4 e1 = reinterpret_cast<const float4*>(eo + (size_t)s1 * kD)[d4];
  float4 o;
  o.x = xv.x + e0.x + e1.x;
  o.y = xv.y + e0.y + e1.y;
  o.z = xv.z + e0.z + e1.z;
  o.w = xv.w + e0.w + e1.w;
  reinterpret_cast<float4*>(out)[idx] = o;
}

}  // namespace

extern "C" void kernel_launch(void* const* d_in, const int* in_sizes, int n_in,
                              void* d_out, int out_size, void* d_ws, size_t ws_size,
                              hipStream_t stream) {
  (void)in_sizes; (void)n_in; (void)out_size; (void)ws_size;
  const float* x      = (const float*)d_in[0];
  const float* ln_w   = (const float*)d_in[1];
  const float* ln_b   = (const float*)d_in[2];
  const float* gate_w = (const float*)d_in[3];
  const float* w1     = (const float*)d_in[4];
  const float* w2     = (const float*)d_in[5];
  float* out  = (float*)d_out;
  float* mask = out + (size_t)kN * kD;   // router_mask appended after moe_out+x

  // Workspace carve (~139 MB, 256B aligned).
  char* p = (char*)d_ws;
  auto carve = [&](size_t bytes) { char* r = p; p += (bytes + 255) & ~(size_t)255; return r; };
  unsigned short* xn   = (unsigned short*)carve((size_t)kN * kD * 2);
  unsigned short* act  = (unsigned short*)carve((size_t)kRows * kH * 2);
  float*          eo   = (float*)carve((size_t)kRows * kD * 4);
  unsigned short* pw1  = (unsigned short*)carve((size_t)kE * kD * (2 * kH) * 2);
  unsigned short* pw2  = (unsigned short*)carve((size_t)kE * kH * kD * 2);
  int*   counts  = (int*)carve(kE * 4);
  int*   offsets = (int*)carve((kE + 1) * 4);
  int*   cursors = (int*)carve(kE * 4);
  int*   sel     = (int*)carve((size_t)kN * 2 * 4);
  float* selw    = (float*)carve((size_t)kN * 2 * 4);
  int*   rowTok  = (int*)carve((size_t)kRows * 4);
  float* rowGate = (float*)carve((size_t)kRows * 4);
  int*   tokSlot = (int*)carve((size_t)kN * 2 * 4);

  k_init<<<1, 64, 0, stream>>>(counts);
  k_ln_router<<<kN, 256, 0, stream>>>(x, ln_w, ln_b, gate_w, xn, mask, counts, sel, selw);
  k_offsets<<<1, 32, 0, stream>>>(counts, offsets, cursors);
  k_scatter<<<kN / 256, 256, 0, stream>>>(sel, selw, cursors, rowTok, rowGate, tokSlot);

  {
    const long total = (long)kE * (2 * kH / 16) * (kD / 32) * 32;
    k_pack<<<(unsigned)((total + 255) / 256), 256, 0, stream>>>(
        w1, pw1, kD, 2 * kH, 2 * kH / 16, kD / 32, total);
  }
  {
    const long total = (long)kE * (kD / 16) * (kH / 32) * 32;
    k_pack<<<(unsigned)((total + 255) / 256), 256, 0, stream>>>(
        w2, pw2, kH, kD, kD / 16, kH / 32, total);
  }

  k_ffn1<<<dim3(kH / 256, kN / 32, kE), 256, 0, stream>>>(xn, pw1, counts, offsets, rowTok, act);
  k_ffn2<<<dim3(kD / 512, kN / 32, kE), 256, 0, stream>>>(act, pw2, counts, offsets, rowGate, eo);
  k_final<<<(kN * kD / 4) / 256, 256, 0, stream>>>(x, eo, tokSlot, out);
}
